// ShapeEncoder_37907381354812
// MI455X (gfx1250) — compile-verified, matrix operand order checked by reading, not run on hardware
//
#include <hip/hip_runtime.h>
#include <hip/hip_bf16.h>

#define CDIV(a,b) (((a)+(b)-1)/(b))
#define BATCH 32
#define NPTS 2048

typedef __attribute__((ext_vector_type(16))) _Float16 v16h;
typedef __attribute__((ext_vector_type(8)))  float    v8f;

// D = A(16x32 f16) * B(32x16 f16) + C(16x16 f32), wave32
__device__ inline v8f wmma16(v16h a, v16h b, v8f c) {
  return __builtin_amdgcn_wmma_f32_16x16x32_f16(false, a, false, b, (short)0, c, false, false);
}
// A-fragment loader: row-major f16 weights, wave32 16-bit A layout (ISA 7.12.2)
__device__ inline void loadA(v16h& a, const _Float16* wp, int half) {
#pragma unroll
  for (int v = 0; v < 4; ++v) {
    int k0 = half * 8 + 2 * v, k1 = k0 + 16;
    a[2 * v] = wp[k0];     a[2 * v + 1] = wp[k0 + 1];
    a[8 + 2 * v] = wp[k1]; a[9 + 2 * v] = wp[k1 + 1];
  }
}

// ---------------------------------------------------------------- utility
__global__ void kfill(float* p, float v, long n) {
  long i = (long)blockIdx.x * blockDim.x + threadIdx.x;
  if (i < n) p[i] = v;
}
__global__ void kadd(float* dst, const float* src, long n) {
  long i = (long)blockIdx.x * blockDim.x + threadIdx.x;
  if (i < n) dst[i] += src[i];
}
// pack f32 (rows x ksrc) -> f16 (rows x kpad), zero padded
__global__ void kpack(const float* src, _Float16* dst, int rows, int ksrc, int kpad) {
  long i = (long)blockIdx.x * blockDim.x + threadIdx.x;
  if (i >= (long)rows * kpad) return;
  int r = (int)(i / kpad), k = (int)(i % kpad);
  dst[i] = (k < ksrc) ? (_Float16)src[(long)r * ksrc + k] : (_Float16)0.f;
}

// ---------------------------------------------------------------- voxelize
__global__ void vox_stats(const float* coords, float* mean3, float* scale, int Np) {
  int b = blockIdx.x, tid = threadIdx.x;
  const float* c = coords + (long)b * 3 * Np;
  __shared__ float red[256];
  __shared__ float m[3];
  for (int d = 0; d < 3; ++d) {
    float s = 0.f;
    for (int n = tid; n < Np; n += 256) s += c[(long)d * Np + n];
    red[tid] = s; __syncthreads();
    for (int st = 128; st > 0; st >>= 1) { if (tid < st) red[tid] += red[tid + st]; __syncthreads(); }
    if (tid == 0) m[d] = red[0] / (float)Np;
    __syncthreads();
  }
  if (tid < 3) mean3[b * 3 + tid] = m[tid];
  float mx = 0.f;
  for (int n = tid; n < Np; n += 256) {
    float dx = c[n] - m[0], dy = c[Np + n] - m[1], dz = c[2 * Np + n] - m[2];
    mx = fmaxf(mx, dx * dx + dy * dy + dz * dz);
  }
  red[tid] = mx; __syncthreads();
  for (int st = 128; st > 0; st >>= 1) { if (tid < st) red[tid] = fmaxf(red[tid], red[tid + st]); __syncthreads(); }
  if (tid == 0) scale[b] = sqrtf(red[0]) * 2.0f;
}

__global__ void vox_nc(const float* coords, const float* mean3, const float* scale,
                       float* nc, int* flat, int Np, int R) {
  long i = (long)blockIdx.x * blockDim.x + threadIdx.x;
  if (i >= (long)BATCH * Np) return;
  int b = (int)(i / Np), n = (int)(i % Np);
  float s = scale[b]; if (s == 0.f) s = 1.f;
  int vi[3];
  for (int c = 0; c < 3; ++c) {
    float v = coords[((long)b * 3 + c) * Np + n];
    float f = ((v - mean3[b * 3 + c]) / s + 0.5f) * (float)R;
    f = fminf(fmaxf(f, 0.f), (float)R - 1.f);
    nc[((long)b * 3 + c) * Np + n] = f;
    vi[c] = (int)rintf(f);
  }
  flat[(long)b * Np + n] = (vi[0] * R + vi[1]) * R + vi[2];
}

__global__ void vox_scatter(const float* feat, const int* flat, float* grid, float* cnt,
                            int Cin, int V, int Np) {
  long i = (long)blockIdx.x * blockDim.x + threadIdx.x;
  if (i >= (long)BATCH * Np) return;
  int b = (int)(i / Np), n = (int)(i % Np);
  int idx = flat[(long)b * Np + n];
  atomicAdd(&cnt[(long)b * V + idx], 1.f);
  for (int c = 0; c < Cin; ++c)
    atomicAdd(&grid[((long)b * Cin + c) * V + idx], feat[((long)b * Cin + c) * Np + n]);
}

__global__ void vox_div(float* grid, const float* cnt, int Cin, int V) {
  long i = (long)blockIdx.x * blockDim.x + threadIdx.x;
  if (i >= (long)BATCH * Cin * V) return;
  int b = (int)(i / ((long)Cin * V));
  int v = (int)(i % V);
  grid[i] /= fmaxf(cnt[(long)b * V + v], 1.f);
}

// ---------------------------------------------------------------- 3D conv: implicit GEMM w/ WMMA
// Cout == 32 fixed: one wave computes two 16(cout) x 16(voxel) tiles sharing the B fragment.
__global__ void conv3d_wmma(const float* __restrict__ X, const _Float16* __restrict__ W,
                            const float* __restrict__ bias, float* __restrict__ Y,
                            int Cin, int R, int Kpad) {
  int V = R * R * R;
  int lane = threadIdx.x, half = lane >> 4, lr = lane & 15;
  int vj = blockIdx.x * 16 + lr;          // voxel column (N)
  int b = blockIdx.y;
  int wz = vj % R, wy = (vj / R) % R, wx = vj / (R * R);
  const float* xb = X + (long)b * Cin * V;

  auto fill8 = [&](v16h& bv, int base, int ci, int kid) {
    int dd = kid / 9, r9 = kid - dd * 9;
    int dh = r9 / 3, dw = r9 - dh * 3;
    int zx = wx + dd - 1, zy = wy + dh - 1, zz = wz + dw - 1;
#pragma unroll
    for (int t = 0; t < 8; ++t) {
      float v = 0.f;
      if (ci < Cin && (unsigned)zx < (unsigned)R && (unsigned)zy < (unsigned)R &&
          (unsigned)zz < (unsigned)R)
        v = xb[(long)ci * V + ((long)zx * R + zy) * R + zz];
      bv[base + t] = (_Float16)v;
      ++dw; ++zz;
      if (dw == 3) {
        dw = 0; zz = wz - 1; ++dh; ++zy;
        if (dh == 3) {
          dh = 0; zy = wy - 1; ++dd; ++zx;
          if (dd == 3) { dd = 0; zx = wx - 1; ++ci; }
        }
      }
    }
  };

  v8f acc0 = {}, acc1 = {};
  int ci0 = 0, kid0 = half * 8;
  int ci1 = 0, kid1 = 16 + half * 8;
  for (int kb = 0; kb < Kpad; kb += 32) {
    v16h a0, a1, bv;
    loadA(a0, W + (long)lr * Kpad + kb, half);
    loadA(a1, W + (long)(lr + 16) * Kpad + kb, half);
    fill8(bv, 0, ci0, kid0);
    fill8(bv, 8, ci1, kid1);
    acc0 = wmma16(a0, bv, acc0);
    acc1 = wmma16(a1, bv, acc1);
    ci0 += 1; kid0 += 5; if (kid0 >= 27) { kid0 -= 27; ci0 += 1; }
    ci1 += 1; kid1 += 5; if (kid1 >= 27) { kid1 -= 27; ci1 += 1; }
  }
#pragma unroll
  for (int r = 0; r < 8; ++r) {
    int m = r + 8 * half;
    Y[((long)b * 32 + m) * V + vj] = acc0[r] + bias[m];
    Y[((long)b * 32 + m + 16) * V + vj] = acc1[r] + bias[m + 16];
  }
}

// ---------------------------------------------------------------- generic strided GEMM w/ WMMA
// Msz % 32 == 0, grid.y = Msz/32 (two M tiles per wave share the B fragment).
__global__ void gemm_wmma(const _Float16* __restrict__ W, int Kpad, int Ksrc,
                          const float* __restrict__ Bsrc, long long sK, long long sN, long long sB,
                          const float* __restrict__ bias,
                          float* __restrict__ C, long long sCm, long long sCn, long long sCb,
                          int Msz, int Nsz) {
  int lane = threadIdx.x, half = lane >> 4, lr = lane & 15;
  int n = blockIdx.x * 16 + lr;
  int m0 = blockIdx.y * 32 + lr;
  long long z = blockIdx.z;
  const float* bp = Bsrc + z * sB;
  bool nok = n < Nsz;
  v8f acc0 = {}, acc1 = {};
  for (int kb = 0; kb < Kpad; kb += 32) {
    v16h a0, a1, bv;
    loadA(a0, W + (long long)m0 * Kpad + kb, half);
    loadA(a1, W + (long long)(m0 + 16) * Kpad + kb, half);
#pragma unroll
    for (int t = 0; t < 8; ++t) {
      int ka = kb + half * 8 + t, kc = ka + 16;
      bv[t] = (nok && ka < Ksrc) ? (_Float16)bp[(long long)ka * sK + (long long)n * sN]
                                 : (_Float16)0.f;
      bv[8 + t] = (nok && kc < Ksrc) ? (_Float16)bp[(long long)kc * sK + (long long)n * sN]
                                     : (_Float16)0.f;
    }
    acc0 = wmma16(a0, bv, acc0);
    acc1 = wmma16(a1, bv, acc1);
  }
  if (nok) {
    float* cp = C + z * sCb;
#pragma unroll
    for (int r = 0; r < 8; ++r) {
      int m = blockIdx.y * 32 + r + 8 * half;
      if (m < Msz) cp[(long long)m * sCm + (long long)n * sCn] = acc0[r] + bias[m];
      int m2 = m + 16;
      if (m2 < Msz) cp[(long long)m2 * sCm + (long long)n * sCn] = acc1[r] + bias[m2];
    }
  }
}

// split-K WMMA GEMM: grid.z = K-slices; partials P[s][m][n] (no bias)
__global__ void gemm_splitk(const _Float16* __restrict__ W, int Kpad, int Ksrc, int Kchunk,
                            const float* __restrict__ Bsrc, long long sK, long long sN,
                            float* __restrict__ P, int Msz, int Nsz) {
  int lane = threadIdx.x, half = lane >> 4, lr = lane & 15;
  int n = blockIdx.x * 16 + lr;
  int m0 = blockIdx.y * 32 + lr;
  int sidx = blockIdx.z;
  int ks = sidx * Kchunk;
  int ke = ks + Kchunk; if (ke > Kpad) ke = Kpad;
  bool nok = n < Nsz;
  v8f acc0 = {}, acc1 = {};
  for (int kb = ks; kb < ke; kb += 32) {
    v16h a0, a1, bv;
    loadA(a0, W + (long long)m0 * Kpad + kb, half);
    loadA(a1, W + (long long)(m0 + 16) * Kpad + kb, half);
#pragma unroll
    for (int t = 0; t < 8; ++t) {
      int ka = kb + half * 8 + t, kc = ka + 16;
      bv[t] = (nok && ka < Ksrc) ? (_Float16)Bsrc[(long long)ka * sK + (long long)n * sN]
                                 : (_Float16)0.f;
      bv[8 + t] = (nok && kc < Ksrc) ? (_Float16)Bsrc[(long long)kc * sK + (long long)n * sN]
                                     : (_Float16)0.f;
    }
    acc0 = wmma16(a0, bv, acc0);
    acc1 = wmma16(a1, bv, acc1);
  }
  if (nok) {
    float* cp = P + (long)sidx * Msz * Nsz;
#pragma unroll
    for (int r = 0; r < 8; ++r) {
      int m = blockIdx.y * 32 + r + 8 * half;
      cp[(long)m * Nsz + n] = acc0[r];
      cp[(long)(m + 16) * Nsz + n] = acc1[r];
    }
  }
}
__global__ void splitk_reduce(const float* P, const float* bias, float* C,
                              long long sCm, long long sCn, int S, int Msz, int Nsz) {
  long i = (long)blockIdx.x * blockDim.x + threadIdx.x;
  if (i >= (long)Msz * Nsz) return;
  int m = (int)(i / Nsz), n = (int)(i % Nsz);
  float acc = bias[m];
  for (int s = 0; s < S; ++s) acc += P[((long)s * Msz + m) * Nsz + n];
  C[(long)m * sCm + (long)n * sCn] = acc;
}

// SA layer-1 GEMM with fused neighbor gather (O % 32 == 0, grid.y = O/32)
__global__ void gemm_sa(const _Float16* __restrict__ W, int Kpad, int Ksrc,
                        const float* __restrict__ coords, const float* __restrict__ feats,
                        const float* __restrict__ centers, const int* __restrict__ nidx,
                        const float* __restrict__ bias, float* __restrict__ C,
                        int O, int Mq, int Kq, int Np, int Cf) {
  int lane = threadIdx.x, half = lane >> 4, lr = lane & 15;
  long L = (long)Mq * Kq;
  int n = blockIdx.x * 16 + lr;
  int o0 = blockIdx.y * 32 + lr;
  int b = blockIdx.z;
  int m = n / Kq;
  int idx = nidx[((long)b * Mq + m) * Kq + (n - m * Kq)];
  if (idx >= Np) idx = Np - 1;
  auto fetchB = [&](int c) -> _Float16 {
    if (c >= Ksrc) return (_Float16)0.f;
    float v;
    if (c < 3) v = coords[((long)b * 3 + c) * Np + idx] - centers[((long)b * 3 + c) * Mq + m];
    else v = feats[((long)b * Cf + (c - 3)) * Np + idx];
    return (_Float16)v;
  };
  v8f acc0 = {}, acc1 = {};
  for (int kb = 0; kb < Kpad; kb += 32) {
    v16h a0, a1, bv;
    loadA(a0, W + (long)o0 * Kpad + kb, half);
    loadA(a1, W + (long)(o0 + 16) * Kpad + kb, half);
#pragma unroll
    for (int t = 0; t < 8; ++t) {
      bv[t] = fetchB(kb + half * 8 + t);
      bv[8 + t] = fetchB(kb + 16 + half * 8 + t);
    }
    acc0 = wmma16(a0, bv, acc0);
    acc1 = wmma16(a1, bv, acc1);
  }
#pragma unroll
  for (int r = 0; r < 8; ++r) {
    int mo = blockIdx.y * 32 + r + 8 * half;
    C[((long)b * O + mo) * L + n] = acc0[r] + bias[mo];
    C[((long)b * O + mo + 16) * L + n] = acc1[r] + bias[mo + 16];
  }
}

// ---------------------------------------------------------------- batch norm (two-stage, deterministic)
__global__ void bn_part(const float* x, float* part, int Bc, int C, long L, int S) {
  int c = blockIdx.x, s = blockIdx.y, tid = threadIdx.x;
  long tot = (long)Bc * L;
  long per = (tot + S - 1) / S;
  long lo = (long)s * per, hi = lo + per;
  if (hi > tot) hi = tot;
  float sum = 0.f, sum2 = 0.f;
  for (long i = lo + tid; i < hi; i += 256) {
    long b = i / L, l = i % L;
    float v = x[((long)b * C + c) * L + l];
    sum += v; sum2 += v * v;
  }
  __shared__ float rs[256], rs2[256];
  rs[tid] = sum; rs2[tid] = sum2; __syncthreads();
  for (int st = 128; st > 0; st >>= 1) {
    if (tid < st) { rs[tid] += rs[tid + st]; rs2[tid] += rs2[tid + st]; }
    __syncthreads();
  }
  if (tid == 0) {
    part[((long)c * S + s) * 2 + 0] = rs[0];
    part[((long)c * S + s) * 2 + 1] = rs2[0];
  }
}
__global__ void bn_final(const float* part, float* mean, float* rstd, int S, long tot, float eps) {
  int c = blockIdx.x, tid = threadIdx.x;  // 64 threads
  __shared__ float rs[64], rs2[64];
  float s1 = 0.f, s2 = 0.f;
  for (int s = tid; s < S; s += 64) {
    s1 += part[((long)c * S + s) * 2 + 0];
    s2 += part[((long)c * S + s) * 2 + 1];
  }
  rs[tid] = s1; rs2[tid] = s2; __syncthreads();
  for (int st = 32; st > 0; st >>= 1) {
    if (tid < st) { rs[tid] += rs[tid + st]; rs2[tid] += rs2[tid + st]; }
    __syncthreads();
  }
  if (tid == 0) {
    float m = rs[0] / (float)tot;
    float var = rs2[0] / (float)tot - m * m;
    mean[c] = m;
    rstd[c] = rsqrtf(var + eps);
  }
}
// act: 0=none 1=relu 2=leaky(0.1)
__global__ void bn_apply(float* x, const float* mean, const float* rstd,
                         const float* g, const float* bt, int Bc, int C, long L, int act) {
  long i = (long)blockIdx.x * blockDim.x + threadIdx.x;
  if (i >= (long)Bc * C * L) return;
  long rem = i % ((long)C * L);
  int c = (int)(rem / L);
  float v = (x[i] - mean[c]) * rstd[c] * g[c] + bt[c];
  if (act == 1) v = fmaxf(v, 0.f);
  else if (act == 2) v = (v >= 0.f) ? v : 0.1f * v;
  x[i] = v;
}

// ---------------------------------------------------------------- SE block
__global__ void se_mean(const float* x, float* s, int C, int V) {
  int bc = blockIdx.x, tid = threadIdx.x;
  __shared__ float red[256];
  const float* p = x + (long)bc * V;
  float acc = 0.f;
  for (int v = tid; v < V; v += 256) acc += p[v];
  red[tid] = acc; __syncthreads();
  for (int st = 128; st > 0; st >>= 1) { if (tid < st) red[tid] += red[tid + st]; __syncthreads(); }
  if (tid == 0) s[bc] = red[0] / (float)V;
}
__global__ void se_fc(const float* s, const float* w1, const float* b1,
                      const float* w2, const float* b2, float* out, int C, int H) {
  int b = blockIdx.x, tid = threadIdx.x;
  __shared__ float ss[64], sh[16];
  ss[tid] = s[(long)b * C + tid];
  __syncthreads();
  if (tid < H) {
    float h = b1[tid];
    for (int c = 0; c < C; ++c) h += w1[(long)tid * C + c] * ss[c];
    sh[tid] = fmaxf(h, 0.f);
  }
  __syncthreads();
  float o = b2[tid];
  for (int j = 0; j < H; ++j) o += w2[(long)tid * H + j] * sh[j];
  out[(long)b * C + tid] = 1.f / (1.f + expf(-o));
}
__global__ void se_scale(float* x, const float* s, int C, int V) {
  long i = (long)blockIdx.x * blockDim.x + threadIdx.x;
  if (i >= (long)BATCH * C * V) return;
  int bc = (int)(i / V);
  x[i] *= s[bc];
}

// ---------------------------------------------------------------- devoxelize (trilinear)
__global__ void devox(const float* grid, const float* nc, float* out, int C, int R, int Np) {
  long i = (long)blockIdx.x * blockDim.x + threadIdx.x;
  if (i >= (long)BATCH * C * Np) return;
  int b = (int)(i / ((long)C * Np));
  long rem = i % ((long)C * Np);
  int c = (int)(rem / Np), n = (int)(rem % Np);
  int V = R * R * R;
  float fx = nc[((long)b * 3 + 0) * Np + n];
  float fy = nc[((long)b * 3 + 1) * Np + n];
  float fz = nc[((long)b * 3 + 2) * Np + n];
  int x0 = (int)floorf(fx), y0 = (int)floorf(fy), z0 = (int)floorf(fz);
  int x1 = min(x0 + 1, R - 1), y1 = min(y0 + 1, R - 1), z1 = min(z0 + 1, R - 1);
  float rx = fx - x0, ry = fy - y0, rz = fz - z0;
  const float* g = grid + ((long)b * C + c) * V;
  float acc = 0.f;
  for (int dx = 0; dx < 2; ++dx)
    for (int dy = 0; dy < 2; ++dy)
      for (int dz = 0; dz < 2; ++dz) {
        int ix = dx ? x1 : x0, iy = dy ? y1 : y0, iz = dz ? z1 : z0;
        float w = (dx ? rx : 1.f - rx) * (dy ? ry : 1.f - ry) * (dz ? rz : 1.f - rz);
        acc += w * g[((long)ix * R + iy) * R + iz];
      }
  out[((long)b * C + c) * Np + n] = acc;
}

// ---------------------------------------------------------------- FPS (wave32 shuffle argmax)
__global__ void fps_kernel(const float* coords, int* out, int Np, int M) {
  int b = blockIdx.x, tid = threadIdx.x;
  const float* cx = coords + (long)b * 3 * Np;
  float d[8];
  int ppt = CDIV(Np, 256);
  for (int j = 0; j < 8; ++j) d[j] = 1e10f;
  __shared__ float sv[8];
  __shared__ int si[8];
  __shared__ int s_last;
  __shared__ float lp[3];
  if (tid == 0) s_last = 0;
  __syncthreads();
  for (int it = 0; it < M; ++it) {
    int last = s_last;
    if (tid == 0) out[(long)b * M + it] = last;
    if (tid < 3) lp[tid] = cx[(long)tid * Np + last];
    __syncthreads();
    float bm = -1.f; int bi = 0;
    for (int j = 0; j < ppt; ++j) {
      int n = tid + j * 256;
      if (n < Np) {
        float dx = cx[n] - lp[0], dy = cx[Np + n] - lp[1], dz = cx[2 * Np + n] - lp[2];
        float dd = dx * dx + dy * dy + dz * dz;
        d[j] = fminf(d[j], dd);
        if (d[j] > bm) { bm = d[j]; bi = n; }
      }
    }
    // intra-wave argmax: 5 shuffle butterflies, no barrier
#pragma unroll
    for (int off = 16; off > 0; off >>= 1) {
      float ov = __shfl_xor(bm, off, 32);
      int oi = __shfl_xor(bi, off, 32);
      if (ov > bm) { bm = ov; bi = oi; }
    }
    if ((tid & 31) == 0) { sv[tid >> 5] = bm; si[tid >> 5] = bi; }
    __syncthreads();
    if (tid < 8) {
      float v2 = sv[tid]; int i2 = si[tid];
#pragma unroll
      for (int off = 4; off > 0; off >>= 1) {
        float ov = __shfl_xor(v2, off, 32);
        int oi = __shfl_xor(i2, off, 32);
        if (ov > v2) { v2 = ov; i2 = oi; }
      }
      if (tid == 0) s_last = i2;
    }
    __syncthreads();
  }
}
__global__ void gather_centers(const float* coords, const int* idx, float* centers, int Np, int M) {
  long i = (long)blockIdx.x * blockDim.x + threadIdx.x;
  if (i >= (long)BATCH * 3 * M) return;
  int b = (int)(i / (3 * M));
  long rem = i % (3 * M);
  int c = (int)(rem / M), m = (int)(rem % M);
  centers[((long)b * 3 + c) * M + m] = coords[((long)b * 3 + c) * Np + idx[(long)b * M + m]];
}
__global__ void ballq(const float* coords, const float* centers, int* nidx,
                      int Np, int M, int K, float r2) {
  long i = (long)blockIdx.x * blockDim.x + threadIdx.x;
  if (i >= (long)BATCH * M) return;
  int b = (int)(i / M), m = (int)(i % M);
  float cx = centers[((long)b * 3 + 0) * M + m];
  float cy = centers[((long)b * 3 + 1) * M + m];
  float cz = centers[((long)b * 3 + 2) * M + m];
  const float* p = coords + (long)b * 3 * Np;
  int first = 0, cnt = 0;
  bool has = false;
  int* q = nidx + i * K;
  for (int j = 0; j < Np; ++j) {
    float dx = p[j] - cx, dy = p[Np + j] - cy, dz = p[2 * Np + j] - cz;
    if (dx * dx + dy * dy + dz * dz < r2) {
      if (!has) { first = j; has = true; }
      if (cnt < K) q[cnt] = j;
      cnt++;
    }
  }
  for (int t = cnt; t < K; ++t) q[t] = first;
}
__global__ void maxk(const float* x, float* out, int O, int Mq, int K) {
  long i = (long)blockIdx.x * blockDim.x + threadIdx.x;
  if (i >= (long)BATCH * O * Mq) return;
  long bo = i / Mq;
  int m = (int)(i % Mq);
  const float* p = x + bo * (long)Mq * K + (long)m * K;
  float mx = p[0];
  for (int k = 1; k < K; ++k) mx = fmaxf(mx, p[k]);
  out[i] = mx;
}

// ================================================================ host side
struct PvP { const float *b1,*b2,*c1b,*c1w,*c2b,*c2w,*g1,*g2,*pb,*pbb,*pg,*pw,*seb1,*seb2,*sew1,*sew2; };
struct HdP { const float *b1,*b2,*b3,*bb1,*bb2,*g1,*g2,*w1,*w2,*w3; };
struct Ws {
  float *g0,*g1,*cnt,*nc,*mean3,*scale,*bnm,*bnr,*bnpart,*ses,*fa,*fb,*ptb,*cen1,*cen2,*clsout,*hy1,*hy2,*skp;
  int *flat,*fpsIdx,*nidx;
  _Float16* wpk;
};

static PvP loadPv(void* const* d, int k) {
  PvP p;
  p.b1=(const float*)d[k+0];  p.b2=(const float*)d[k+1];  p.c1b=(const float*)d[k+2];
  p.c1w=(const float*)d[k+3]; p.c2b=(const float*)d[k+4]; p.c2w=(const float*)d[k+5];
  p.g1=(const float*)d[k+6];  p.g2=(const float*)d[k+7];  p.pb=(const float*)d[k+8];
  p.pbb=(const float*)d[k+9]; p.pg=(const float*)d[k+10]; p.pw=(const float*)d[k+11];
  p.seb1=(const float*)d[k+12]; p.seb2=(const float*)d[k+13];
  p.sew1=(const float*)d[k+14]; p.sew2=(const float*)d[k+15];
  return p;
}
static HdP loadHd(void* const* d, int k) {
  HdP h;
  h.b1=(const float*)d[k+0];  h.b2=(const float*)d[k+1];  h.b3=(const float*)d[k+2];
  h.bb1=(const float*)d[k+3]; h.bb2=(const float*)d[k+4]; h.g1=(const float*)d[k+5];
  h.g2=(const float*)d[k+6];  h.w1=(const float*)d[k+7];  h.w2=(const float*)d[k+8];
  h.w3=(const float*)d[k+9];
  return h;
}

// two-stage BN + activation
static void run_bn(float* x, int Bc, int C, long L, const float* g, const float* bt,
                   int act, float eps, Ws& w, hipStream_t s) {
  long tot = (long)Bc * L;
  int S = (int)((tot + 65535) / 65536);
  if (S > 64) S = 64;
  if (S < 1) S = 1;
  bn_part<<<dim3(C, S), 256, 0, s>>>(x, w.bnpart, Bc, C, L, S);
  bn_final<<<C, 64, 0, s>>>(w.bnpart, w.bnm, w.bnr, S, tot, eps);
  bn_apply<<<(int)CDIV((long)Bc * C * L, 256), 256, 0, s>>>(x, w.bnm, w.bnr, g, bt, Bc, C, L, act);
}

static void run_pvconv(const float* feat, const float* coords, float* out,
                       int Cin, int R, int Np, const PvP& P, Ws& w, hipStream_t s) {
  long V = (long)R * R * R;
  // voxelize
  vox_stats<<<BATCH, 256, 0, s>>>(coords, w.mean3, w.scale, Np);
  vox_nc<<<CDIV(BATCH * Np, 256), 256, 0, s>>>(coords, w.mean3, w.scale, w.nc, w.flat, Np, R);
  kfill<<<CDIV(BATCH * Cin * V, 256), 256, 0, s>>>(w.g0, 0.f, (long)BATCH * Cin * V);
  kfill<<<CDIV(BATCH * V, 256), 256, 0, s>>>(w.cnt, 0.f, (long)BATCH * V);
  vox_scatter<<<CDIV(BATCH * Np, 256), 256, 0, s>>>(feat, w.flat, w.g0, w.cnt, Cin, (int)V, Np);
  vox_div<<<CDIV(BATCH * Cin * V, 256), 256, 0, s>>>(w.g0, w.cnt, Cin, (int)V);
  // conv1 + bn + leaky
  int Kp1 = CDIV(Cin * 27, 32) * 32;
  kpack<<<CDIV(32 * Kp1, 256), 256, 0, s>>>(P.c1w, w.wpk, 32, Cin * 27, Kp1);
  conv3d_wmma<<<dim3((int)(V / 16), BATCH), 32, 0, s>>>(w.g0, w.wpk, P.c1b, w.g1, Cin, R, Kp1);
  run_bn(w.g1, BATCH, 32, V, P.g1, P.b1, 2, 1e-4f, w, s);
  // conv2 + bn + leaky
  kpack<<<CDIV(32 * 864, 256), 256, 0, s>>>(P.c2w, w.wpk, 32, 864, 864);
  conv3d_wmma<<<dim3((int)(V / 16), BATCH), 32, 0, s>>>(w.g1, w.wpk, P.c2b, w.g0, 32, R, 864);
  run_bn(w.g0, BATCH, 32, V, P.g2, P.b2, 2, 1e-4f, w, s);
  // SE
  se_mean<<<BATCH * 32, 256, 0, s>>>(w.g0, w.ses, 32, (int)V);
  se_fc<<<BATCH, 32, 0, s>>>(w.ses, P.sew1, P.seb1, P.sew2, P.seb2, w.ses, 32, 4);
  se_scale<<<CDIV(BATCH * 32 * V, 256), 256, 0, s>>>(w.g0, w.ses, 32, (int)V);
  // devoxelize
  devox<<<CDIV(BATCH * 32 * Np, 256), 256, 0, s>>>(w.g0, w.nc, out, 32, R, Np);
  // point transform + bn + relu, add
  kpack<<<CDIV(32 * 32, 256), 256, 0, s>>>(P.pw, w.wpk, 32, Cin, 32);
  gemm_wmma<<<dim3(Np / 16, 1, BATCH), 32, 0, s>>>(w.wpk, 32, Cin, feat, Np, 1, (long long)Cin * Np,
      P.pb, w.ptb, Np, 1, (long long)32 * Np, 32, Np);
  run_bn(w.ptb, BATCH, 32, Np, P.pg, P.pbb, 1, 1e-4f, w, s);
  kadd<<<CDIV(BATCH * 32 * Np, 256), 256, 0, s>>>(out, w.ptb, (long)BATCH * 32 * Np);
}

static void run_sa(const float* feat, const float* coords, int Np, int Cf,
                   int M, int K, float radius, void* const* d_in, int sb, int O1, int O2,
                   float* outFeat, float* outCen, Ws& w, hipStream_t s) {
  const float *w0=(const float*)d_in[sb+0], *b0=(const float*)d_in[sb+1];
  const float *gg0=(const float*)d_in[sb+2], *bb0=(const float*)d_in[sb+3];
  const float *w1p=(const float*)d_in[sb+4], *b1p=(const float*)d_in[sb+5];
  const float *gg1=(const float*)d_in[sb+6], *bb1=(const float*)d_in[sb+7];
  long L = (long)M * K;
  fps_kernel<<<BATCH, 256, 0, s>>>(coords, w.fpsIdx, Np, M);
  gather_centers<<<CDIV(BATCH * 3 * M, 256), 256, 0, s>>>(coords, w.fpsIdx, outCen, Np, M);
  ballq<<<CDIV(BATCH * M, 256), 256, 0, s>>>(coords, outCen, w.nidx, Np, M, K, radius * radius);
  int Cin = 3 + Cf;
  kpack<<<CDIV(O1 * 64, 256), 256, 0, s>>>(w0, w.wpk, O1, Cin, 64);
  gemm_sa<<<dim3((int)(L / 16), O1 / 32, BATCH), 32, 0, s>>>(w.wpk, 64, Cin, coords, feat, outCen,
      w.nidx, b0, w.g1, O1, M, K, Np, Cf);
  run_bn(w.g1, BATCH, O1, L, gg0, bb0, 1, 1e-5f, w, s);
  kpack<<<CDIV(O2 * 32, 256), 256, 0, s>>>(w1p, w.wpk, O2, O1, 32);
  gemm_wmma<<<dim3((int)(L / 16), O2 / 32, BATCH), 32, 0, s>>>(w.wpk, 32, O1, w.g1, L, 1,
      (long long)O1 * L, b1p, w.g0, L, 1, (long long)O2 * L, O2, (int)L);
  run_bn(w.g0, BATCH, O2, L, gg1, bb1, 1, 1e-5f, w, s);
  maxk<<<CDIV(BATCH * O2 * M, 256), 256, 0, s>>>(w.g0, outFeat, O2, M, K);
}

static void run_head(const HdP& h, const float* clsflat, float* outp, Ws& w, hipStream_t s) {
  // layer 1: 256 x 32 over K=32768 -> split-K across 32 slices (512 waves)
  const int S1 = 32, KCH = 32768 / S1;
  kpack<<<CDIV(256 * 32768, 256), 256, 0, s>>>(h.w1, w.wpk, 256, 32768, 32768);
  gemm_splitk<<<dim3(2, 8, S1), 32, 0, s>>>(w.wpk, 32768, 32768, KCH, clsflat, 1, 32768,
      w.skp, 256, BATCH);
  splitk_reduce<<<CDIV(256 * BATCH, 256), 256, 0, s>>>(w.skp, h.b1, w.hy1, 1, 256, S1, 256, BATCH);
  run_bn(w.hy1, BATCH, 256, 1, h.g1, h.bb1, 1, 1e-5f, w, s);
  // layer 2
  kpack<<<CDIV(256 * 256, 256), 256, 0, s>>>(h.w2, w.wpk, 256, 256, 256);
  gemm_wmma<<<dim3(2, 8, 1), 32, 0, s>>>(w.wpk, 256, 256, w.hy1, 1, 256, 0, h.b2,
      w.hy2, 1, 256, 0, 256, BATCH);
  run_bn(w.hy2, BATCH, 256, 1, h.g2, h.bb2, 1, 1e-5f, w, s);
  // layer 3
  kpack<<<CDIV(128 * 256, 256), 256, 0, s>>>(h.w3, w.wpk, 128, 256, 256);
  gemm_wmma<<<dim3(2, 4, 1), 32, 0, s>>>(w.wpk, 256, 256, w.hy2, 1, 256, 0, h.b3,
      outp, 1, 128, 0, 128, BATCH);
}

extern "C" void kernel_launch(void* const* d_in, const int* in_sizes, int n_in,
                              void* d_out, int out_size, void* d_ws, size_t ws_size,
                              hipStream_t stream) {
  (void)in_sizes; (void)out_size; (void)ws_size;
  if (n_in < 87) return;
  const long V1 = 32768;
  char* base = (char*)d_ws;
  size_t off = 0;
  auto alloc = [&](size_t bytes) -> char* {
    char* p = base + off;
    off = (off + bytes + 255) & ~(size_t)255;
    return p;
  };
  Ws w;
  w.g0     = (float*)alloc((size_t)BATCH * 32 * V1 * 4);   // 128 MB
  w.g1     = (float*)alloc((size_t)BATCH * 32 * V1 * 4);   // 128 MB
  w.cnt    = (float*)alloc((size_t)BATCH * V1 * 4);
  w.nc     = (float*)alloc((size_t)BATCH * 3 * NPTS * 4);
  w.mean3  = (float*)alloc((size_t)BATCH * 3 * 4);
  w.scale  = (float*)alloc((size_t)BATCH * 4);
  w.bnm    = (float*)alloc(256 * 4);
  w.bnr    = (float*)alloc(256 * 4);
  w.bnpart = (float*)alloc((size_t)256 * 64 * 2 * 4);
  w.ses    = (float*)alloc((size_t)BATCH * 32 * 4);
  w.fa     = (float*)alloc((size_t)BATCH * 32 * NPTS * 4);
  w.fb     = (float*)alloc((size_t)BATCH * 32 * NPTS * 4);
  w.ptb    = (float*)alloc((size_t)BATCH * 32 * NPTS * 4);
  w.cen1   = (float*)alloc((size_t)BATCH * 3 * 1024 * 4);
  w.cen2   = (float*)alloc((size_t)BATCH * 3 * 256 * 4);
  w.clsout = (float*)alloc((size_t)BATCH * 128 * 256 * 4);
  w.hy1    = (float*)alloc((size_t)BATCH * 256 * 4);
  w.hy2    = (float*)alloc((size_t)BATCH * 256 * 4);
  w.skp    = (float*)alloc((size_t)32 * 256 * BATCH * 4);  // split-K partials
  w.flat   = (int*)alloc((size_t)BATCH * NPTS * 4);
  w.fpsIdx = (int*)alloc((size_t)BATCH * 1024 * 4);
  w.nidx   = (int*)alloc((size_t)BATCH * 1024 * 32 * 4);
  w.wpk    = (_Float16*)alloc((size_t)256 * 32768 * 2);    // 16 MB pack arena

  // d_in index map (JAX pytree / alphabetical dict order):
  // 0=inputs 1=clsb 2=clsw 3..12=head m 13..28=pv1a 29..44=pv1b 45..60=pv2
  // 61..68=sa1 69..76=sa2 77..86=head v
  const float* inputs = (const float*)d_in[0];
  const float* clsb   = (const float*)d_in[1];
  const float* clsw   = (const float*)d_in[2];
  HdP hm = loadHd(d_in, 3);
  PvP pv1a = loadPv(d_in, 13), pv1b = loadPv(d_in, 29), pv2 = loadPv(d_in, 45);
  HdP hv = loadHd(d_in, 77);
  const float* coords0 = inputs;  // inputs[:, :3, :] == inputs (3 channels)

  run_pvconv(inputs, coords0, w.fa, 3, 32, NPTS, pv1a, w, stream);
  run_pvconv(w.fa, coords0, w.fb, 32, 32, NPTS, pv1b, w, stream);
  run_sa(w.fb, coords0, NPTS, 32, 1024, 32, 0.1f, d_in, 61, 32, 32, w.fa, w.cen1, w, stream);
  run_pvconv(w.fa, w.cen1, w.fb, 32, 16, 1024, pv2, w, stream);
  run_sa(w.fb, w.cen1, 1024, 32, 256, 32, 0.2f, d_in, 69, 32, 64, w.fa, w.cen2, w, stream);

  // cls: (128,64) x (B,64,256) -> (B,128,256) -> flatten (B, 32768)
  kpack<<<CDIV(128 * 64, 256), 256, 0, stream>>>(clsw, w.wpk, 128, 64, 64);
  gemm_wmma<<<dim3(16, 4, BATCH), 32, 0, stream>>>(w.wpk, 64, 64, w.fa, 256, 1,
      (long long)64 * 256, clsb, w.clsout, 256, 1, (long long)128 * 256, 128, 256);

  float* outp = (float*)d_out;
  run_head(hm, w.clsout, outp, w, stream);
  run_head(hv, w.clsout, outp + (size_t)BATCH * 128, w, stream);
}
